// build_rnn_11879879542681
// MI455X (gfx1250) — compile-verified, matrix operand order checked by reading, not run on hardware
//
#include <hip/hip_runtime.h>
#include <hip/hip_bf16.h>

typedef unsigned short u16;
typedef __attribute__((ext_vector_type(16))) __bf16 v16bf;
typedef __attribute__((ext_vector_type(8)))  float  v8f;

#define HL   512          // hidden size H
#define FH   2048         // 4*H
#define BB   64           // batch
#define TT   512          // timesteps
#define DD   5            // input dim
#define EPS  1e-3f

// ---------- helpers ----------
__device__ __forceinline__ u16 f2bf(float f) {
    unsigned u = __float_as_uint(f);
    unsigned r = (u + 0x7FFFu + ((u >> 16) & 1u)) >> 16;   // RNE
    return (u16)r;
}
__device__ __forceinline__ float sigf(float x) { return 1.0f / (1.0f + __expf(-x)); }

struct i4x2 { int4 x; int4 y; };
__device__ __forceinline__ v16bf ld_a_tile(const u16* p) {
    i4x2 t;
    t.x = *reinterpret_cast<const int4*>(p);        // K {koff..koff+7}
    t.y = *reinterpret_cast<const int4*>(p + 16);   // K {koff+16..koff+23}
    return __builtin_bit_cast(v16bf, t);
}
__device__ __forceinline__ v16bf ld_b_tile(const u16* p) {
    i4x2 t;
    t.x = *reinterpret_cast<const int4*>(p);
    t.y = *reinterpret_cast<const int4*>(p + 8);
    return __builtin_bit_cast(v16bf, t);
}

// ---------- weight pre-pack into WMMA B-operand layout ----------
// B tile (K=32 x N=16) stored as 512 contiguous bf16: index = lane*16 + j
// lane<16 : col = nt*16+lane,    j<8 -> k=kc*32+j,     j>=8 -> k=kc*32+16+(j-8)
// lane>=16: col = nt*16+lane-16, j<8 -> k=kc*32+8+j,   j>=8 -> k=kc*32+24+(j-8)
__global__ void pack_b_kernel(const float* __restrict__ W, const float* __restrict__ U,
                              int K0, int K, u16* __restrict__ out) {
    size_t e = (size_t)blockIdx.x * 256 + threadIdx.x;
    size_t total = (size_t)K * FH;
    if (e >= total) return;
    int blk    = (int)(e >> 9);
    int within = (int)(e & 511);
    int lane   = within >> 4;
    int j      = within & 15;
    int kcn    = K >> 5;
    int kc     = blk % kcn;
    int nt     = blk / kcn;
    int half   = lane >> 4;
    int n      = nt * 16 + (lane & 15);
    int kl     = (j < 8) ? (half * 8 + j) : (16 + half * 8 + (j - 8));
    int k      = kc * 32 + kl;
    float v    = (k < K0) ? W[(size_t)k * FH + n] : U[(size_t)(k - K0) * FH + n];
    out[e] = f2bf(v);
}

// ---------- BN folding ----------
__global__ void bn_prep_kernel(const float* g, const float* be, const float* m, const float* v,
                               float* scale, float* shift) {
    int j = blockIdx.x * 256 + threadIdx.x;
    if (j >= HL) return;
    float s = g[j] * rsqrtf(v[j] + EPS);
    scale[j] = s;
    shift[j] = be[j] - m[j] * s;
}

// ---------- state init ----------
__global__ void init_state_kernel(u16* hbf, float* cf, float* hf) {
    int i = blockIdx.x * 256 + threadIdx.x;
    if (i >= BB * HL) return;
    hbf[i] = 0; cf[i] = 0.0f; hf[i] = 0.0f;
}

// ---------- per-step WMMA GEMM: Z[64,2048] = [A0;A1](64,K) x Bpack(K,2048) ----------
// Compile-time K0/K/strides -> all addressing is immediate offsets; the K loop
// is fully unrolled and software-pipelined. Each wave computes a 2Mx2N tile
// block (32x32 output): 4 WMMAs per K chunk off 2 A tiles + 2 B tiles, both
// double-buffered (low register footprint so the allocator keeps the buffers).
template<int K0E, int KE, int SA0, int SA1>
__global__ void gemm_step_kernel(const u16* __restrict__ A0,
                                 const u16* __restrict__ A1,
                                 const u16* __restrict__ Bp,
                                 float* __restrict__ Z) {
    constexpr int kcn   = KE >> 5;
    constexpr int kc0   = K0E >> 5;
    constexpr int bstep = kcn * 512;          // elems between adjacent N tiles

    int wid  = blockIdx.x * (blockDim.x >> 5) + (threadIdx.x >> 5);   // 0..127
    int lane = threadIdx.x & 31;
    int mg   = wid & 1;                 // 2 M groups of 2 tiles (64 rows)
    int ng   = wid >> 1;                // 64 groups of 2 N tiles (2048 cols)
    int half = lane >> 4;
    int lanem = lane & 15;
    int koff = half * 8;
    int mrow0 = mg * 32 + lanem;        // A row, M tile 0
    int mrow1 = mrow0 + 16;             // A row, M tile 1

    const u16* a0p0 = A0 + (size_t)mrow0 * SA0 + koff;
    const u16* a0p1 = A0 + (size_t)mrow1 * SA0 + koff;
    const u16* a1p0 = A1 + (size_t)mrow0 * SA1 + koff;
    const u16* a1p1 = A1 + (size_t)mrow1 * SA1 + koff;
    const u16* bb   = Bp + ((size_t)(ng * 2) * kcn) * 512 + lane * 16;

    v8f acc[2][2];
#pragma unroll
    for (int mi = 0; mi < 2; ++mi)
#pragma unroll
        for (int ni = 0; ni < 2; ++ni) acc[mi][ni] = (v8f){};

    // pipeline prologue (kc = 0 always lies in the A0 range)
    v16bf av[2], bv[2];
    av[0] = ld_a_tile(a0p0);
    av[1] = ld_a_tile(a0p1);
    bv[0] = ld_b_tile(bb);
    bv[1] = ld_b_tile(bb + bstep);

#pragma unroll
    for (int kc = 0; kc < kcn; ++kc) {
        v16bf av_n[2], bv_n[2];
        if (kc + 1 < kcn) {                       // compile-time resolved
            const int kn = kc + 1;
            const u16* p0 = (kn < kc0) ? (a0p0 + kn * 32) : (a1p0 + (kn - kc0) * 32);
            const u16* p1 = (kn < kc0) ? (a0p1 + kn * 32) : (a1p1 + (kn - kc0) * 32);
            av_n[0] = ld_a_tile(p0);
            av_n[1] = ld_a_tile(p1);
            bv_n[0] = ld_b_tile(bb + kn * 512);
            bv_n[1] = ld_b_tile(bb + kn * 512 + bstep);
        }
#pragma unroll
        for (int mi = 0; mi < 2; ++mi)
#pragma unroll
            for (int ni = 0; ni < 2; ++ni)
                acc[mi][ni] = __builtin_amdgcn_wmma_f32_16x16x32_bf16(
                    false, av[mi], false, bv[ni], (short)0, acc[mi][ni], false, false);
        av[0] = av_n[0]; av[1] = av_n[1];
        bv[0] = bv_n[0]; bv[1] = bv_n[1];
    }

    // C layout: VGPR r -> row = r + 8*half ; col = lane&15
#pragma unroll
    for (int mi = 0; mi < 2; ++mi) {
        int row0 = mg * 32 + mi * 16 + half * 8;
#pragma unroll
        for (int ni = 0; ni < 2; ++ni) {
            int n = (ng * 2 + ni) * 16 + lanem;
#pragma unroll
            for (int r = 0; r < 8; ++r)
                Z[(size_t)(row0 + r) * FH + n] = acc[mi][ni][r];
        }
    }
}

// ---------- gate kernels ----------
// layer1: adds b1 + x_t@W1 (K=5), writes seq x1 (f32), BN'd xr (bf16), state h/c
__global__ void gate1_kernel(const float* __restrict__ Z, const float* __restrict__ b1,
                             const float* __restrict__ x, const float* __restrict__ W1,
                             const float* __restrict__ scale, const float* __restrict__ shift,
                             float* __restrict__ cf, u16* __restrict__ hbf,
                             float* __restrict__ x1f, u16* __restrict__ xr, int t) {
    int idx = blockIdx.x * 256 + threadIdx.x;
    if (idx >= BB * HL) return;
    int b = idx >> 9, j = idx & 511;
    const float* xb = x + ((size_t)b * TT + t) * DD;
    float zs[4];
#pragma unroll
    for (int g = 0; g < 4; ++g) {
        int col = g * HL + j;
        float s = Z[(size_t)b * FH + col] + b1[col];
#pragma unroll
        for (int d = 0; d < DD; ++d) s += xb[d] * W1[d * FH + col];
        zs[g] = s;
    }
    float i = sigf(zs[0]), f = sigf(zs[1]), g_ = tanhf(zs[2]), o = sigf(zs[3]);
    float c = f * cf[idx] + i * g_;
    float h = o * tanhf(c);
    cf[idx] = c; hbf[idx] = f2bf(h);
    size_t p = ((size_t)b * TT + t) * HL + j;
    x1f[p] = h;
    xr[p]  = f2bf(h * scale[j] + shift[j]);
}

// layer2: writes residual xs = x1 + h2 (bf16), state h/c
__global__ void gate2_kernel(const float* __restrict__ Z, const float* __restrict__ b2,
                             const float* __restrict__ x1f,
                             float* __restrict__ cf, u16* __restrict__ hbf,
                             u16* __restrict__ xs, int t) {
    int idx = blockIdx.x * 256 + threadIdx.x;
    if (idx >= BB * HL) return;
    int b = idx >> 9, j = idx & 511;
    float zs[4];
#pragma unroll
    for (int g = 0; g < 4; ++g) {
        int col = g * HL + j;
        zs[g] = Z[(size_t)b * FH + col] + b2[col];
    }
    float i = sigf(zs[0]), f = sigf(zs[1]), g_ = tanhf(zs[2]), o = sigf(zs[3]);
    float c = f * cf[idx] + i * g_;
    float h = o * tanhf(c);
    cf[idx] = c; hbf[idx] = f2bf(h);
    size_t p = ((size_t)b * TT + t) * HL + j;
    xs[p] = f2bf(x1f[p] + h);
}

// layer3: only state (final h used by heads)
__global__ void gate3_kernel(const float* __restrict__ Z, const float* __restrict__ b3,
                             float* __restrict__ cf, u16* __restrict__ hbf,
                             float* __restrict__ hf) {
    int idx = blockIdx.x * 256 + threadIdx.x;
    if (idx >= BB * HL) return;
    int b = idx >> 9;
    int j = idx & 511;
    float zs[4];
#pragma unroll
    for (int g = 0; g < 4; ++g) {
        int col = g * HL + j;
        zs[g] = Z[(size_t)b * FH + col] + b3[col];
    }
    float i = sigf(zs[0]), f = sigf(zs[1]), g_ = tanhf(zs[2]), o = sigf(zs[3]);
    float c = f * cf[idx] + i * g_;
    float h = o * tanhf(c);
    cf[idx] = c; hbf[idx] = f2bf(h); hf[idx] = h;
}

// ---------- heads: one block per batch row ----------
__global__ void heads_kernel(const float* __restrict__ hf,
                             const float* d1w, const float* d1b,
                             const float* g2, const float* be2, const float* m2, const float* v2,
                             const float* d2w, const float* d2b,
                             const float* hw, const float* hb,
                             const float* d4w, const float* d4b,
                             const float* d5w, const float* d5b,
                             const float* rw, const float* rb,
                             float* __restrict__ out) {
    __shared__ float sh[HL];
    __shared__ float y1[32], y2[16], z1[128], z2[64];
    int b = blockIdx.x, tid = threadIdx.x;
    for (int k = tid; k < HL; k += 128) sh[k] = hf[(size_t)b * HL + k];
    __syncthreads();
    if (tid < 32) {
        float s = d1b[tid];
        for (int k = 0; k < HL; ++k) s += sh[k] * d1w[k * 32 + tid];
        y1[tid] = (s - m2[tid]) * (g2[tid] * rsqrtf(v2[tid] + EPS)) + be2[tid];
    }
    if (tid < 128) {
        float s = d4b[tid];
        for (int k = 0; k < HL; ++k) s += sh[k] * d4w[k * 128 + tid];
        z1[tid] = s;
    }
    __syncthreads();
    if (tid < 16) {
        float s = d2b[tid];
        for (int k = 0; k < 32; ++k) s += y1[k] * d2w[k * 16 + tid];
        y2[tid] = s;
    }
    if (tid < 64) {
        float s = d5b[tid];
        for (int k = 0; k < 128; ++k) s += z1[k] * d5w[k * 64 + tid];
        z2[tid] = s;
    }
    __syncthreads();
    if (tid == 0) {
        float s = hb[0];
        for (int k = 0; k < 16; ++k) s += y2[k] * hw[k];
        out[b] = sigf(s);
    }
    if (tid < DD) {
        float s = rb[tid];
        for (int k = 0; k < 64; ++k) s += z2[k] * rw[k * DD + tid];
        out[BB + b * DD + tid] = s;
    }
}

// ---------- host ----------
extern "C" void kernel_launch(void* const* d_in, const int* in_sizes, int n_in,
                              void* d_out, int out_size, void* d_ws, size_t ws_size,
                              hipStream_t stream) {
    const float* x   = (const float*)d_in[0];
    const float* W1  = (const float*)d_in[1];
    const float* U1  = (const float*)d_in[2];
    const float* b1  = (const float*)d_in[3];
    const float* g1  = (const float*)d_in[4];
    const float* be1 = (const float*)d_in[5];
    const float* m1  = (const float*)d_in[6];
    const float* v1  = (const float*)d_in[7];
    const float* W2  = (const float*)d_in[8];
    const float* U2  = (const float*)d_in[9];
    const float* b2  = (const float*)d_in[10];
    const float* W3  = (const float*)d_in[11];
    const float* U3  = (const float*)d_in[12];
    const float* b3  = (const float*)d_in[13];
    const float* d1w = (const float*)d_in[14];
    const float* d1b = (const float*)d_in[15];
    const float* g2  = (const float*)d_in[16];
    const float* be2 = (const float*)d_in[17];
    const float* m2  = (const float*)d_in[18];
    const float* v2  = (const float*)d_in[19];
    const float* d2w = (const float*)d_in[20];
    const float* d2b = (const float*)d_in[21];
    const float* hw  = (const float*)d_in[22];
    const float* hb  = (const float*)d_in[23];
    const float* d4w = (const float*)d_in[24];
    const float* d4b = (const float*)d_in[25];
    const float* d5w = (const float*)d_in[26];
    const float* d5b = (const float*)d_in[27];
    const float* rw  = (const float*)d_in[28];
    const float* rb  = (const float*)d_in[29];
    float* out = (float*)d_out;

    // workspace carve
    char* ws = (char*)d_ws;
    size_t off = 0;
    auto carve = [&](size_t bytes) -> char* {
        char* p = ws + off;
        off = (off + bytes + 255) & ~(size_t)255;
        return p;
    };
    u16*   U1p   = (u16*)  carve((size_t)HL  * FH * 2);
    u16*   WU2p  = (u16*)  carve((size_t)2 * HL * FH * 2);
    u16*   WU3p  = (u16*)  carve((size_t)2 * HL * FH * 2);
    float* x1f   = (float*)carve((size_t)BB * TT * HL * 4);
    u16*   xr    = (u16*)  carve((size_t)BB * TT * HL * 2);
    u16*   xs    = (u16*)  carve((size_t)BB * TT * HL * 2);
    float* Zbuf  = (float*)carve((size_t)BB * FH * 4);
    u16*   hbf   = (u16*)  carve((size_t)BB * HL * 2);
    float* cf    = (float*)carve((size_t)BB * HL * 4);
    float* hf    = (float*)carve((size_t)BB * HL * 4);
    float* sc1   = (float*)carve(HL * 4);
    float* sh1   = (float*)carve(HL * 4);

    // one-time (per-call) prep
    pack_b_kernel<<<(HL * FH + 255) / 256, 256, 0, stream>>>(U1, U1, 0, HL, U1p);
    pack_b_kernel<<<(2 * HL * FH + 255) / 256, 256, 0, stream>>>(W2, U2, HL, 2 * HL, WU2p);
    pack_b_kernel<<<(2 * HL * FH + 255) / 256, 256, 0, stream>>>(W3, U3, HL, 2 * HL, WU3p);
    bn_prep_kernel<<<2, 256, 0, stream>>>(g1, be1, m1, v1, sc1, sh1);

    const int gateBlocks = (BB * HL + 255) / 256;   // 128
    const int gemmBlocks = 32;                      // 32 blocks x 4 waves = 128 waves
    const int gemmThreads = 128;

    // ---- LSTM 1 (recurrent-only GEMM, K=512) ----
    init_state_kernel<<<gateBlocks, 256, 0, stream>>>(hbf, cf, hf);
    for (int t = 0; t < TT; ++t) {
        gemm_step_kernel<HL, HL, HL, HL>
            <<<gemmBlocks, gemmThreads, 0, stream>>>(hbf, hbf, U1p, Zbuf);
        gate1_kernel<<<gateBlocks, 256, 0, stream>>>(Zbuf, b1, x, W1, sc1, sh1,
                                                     cf, hbf, x1f, xr, t);
    }
    // ---- LSTM 2 (input xr, K = [xr_t ; h] = 1024) ----
    init_state_kernel<<<gateBlocks, 256, 0, stream>>>(hbf, cf, hf);
    for (int t = 0; t < TT; ++t) {
        gemm_step_kernel<HL, 2 * HL, TT * HL, HL>
            <<<gemmBlocks, gemmThreads, 0, stream>>>(xr + (size_t)t * HL, hbf, WU2p, Zbuf);
        gate2_kernel<<<gateBlocks, 256, 0, stream>>>(Zbuf, b2, x1f, cf, hbf, xs, t);
    }
    // ---- LSTM 3 (input xs = x1 + x2, K = 1024) ----
    init_state_kernel<<<gateBlocks, 256, 0, stream>>>(hbf, cf, hf);
    for (int t = 0; t < TT; ++t) {
        gemm_step_kernel<HL, 2 * HL, TT * HL, HL>
            <<<gemmBlocks, gemmThreads, 0, stream>>>(xs + (size_t)t * HL, hbf, WU3p, Zbuf);
        gate3_kernel<<<gateBlocks, 256, 0, stream>>>(Zbuf, b3, cf, hbf, hf);
    }
    // ---- heads ----
    heads_kernel<<<BB, 128, 0, stream>>>(hf, d1w, d1b, g2, be2, m2, v2, d2w, d2b,
                                         hw, hb, d4w, d4b, d5w, d5b, rw, rb, out);
}
